// GlobalAggregator_28561532518859
// MI455X (gfx1250) — matrix operand order repaired and checked by reading, not compiled, not run on hardware
//
#include <hip/hip_runtime.h>

typedef _Float16 f16;
typedef __attribute__((ext_vector_type(16))) _Float16 v16h;
typedef __attribute__((ext_vector_type(8)))  float    v8f;

#define Bsz 256
#define Nsz 64
#define Ssz 16
#define Dsz 128
// pairs = Bsz*Nsz = 16384, 16 pairs per block -> 1024 blocks

__global__ __launch_bounds__(256) void gagg_kernel(
    const float* __restrict__ selfv,   // [P,128]
    const float* __restrict__ nbrv,    // [P,16,128]
    const float* __restrict__ nbw,     // [P,16]
    const float* __restrict__ w1,      // [129,128]
    const float* __restrict__ w2,      // [128]
    const float* __restrict__ w3,      // [256,128]
    float* __restrict__ out)           // [P,128] out ; then [P,128] nb
{
    // w1p: B-fragment-native swizzle: [(kc*2+half)*128 + col][16 halfs]  (32 KB)
    __shared__ __attribute__((aligned(32))) f16 w1p[8 * Dsz * 16];
    __shared__ f16 w1last[Dsz];                                        // row 128 of w1
    // neighbor tiles (f32) -- reused as swizzled w3 (f16, 64 KB) in stage 3
    __shared__ __attribute__((aligned(32))) float nbuf[8][Ssz * Dsz];  // 64 KB
    __shared__ __attribute__((aligned(16))) float X[16][2 * Dsz];      // 16 KB concat(self,nb)
    __shared__ float wts[16 * Ssz];                                    // 1 KB

    const int tid   = threadIdx.x;
    const int lane  = tid & 31;
    const int wv    = tid >> 5;        // wave 0..7
    const int pbase = blockIdx.x * 16; // first pair of this block
    const int m16   = lane & 15;
    const bool lo   = (lane < 16);
    const int hh    = lo ? 0 : 1;      // which K-half of a 32-chunk this lane's B covers

    // ---- stage 0: cooperative staging (w1 swizzled to fragment layout) ----
    for (int i = tid; i < 128 * Dsz; i += 256) {
        int k = i >> 7, c = i & 127;
        int kc = k >> 5, rem = k & 31, hf = (rem >> 3) & 1, j = (rem & 7) | ((rem >> 4) << 3);
        w1p[((((kc << 1) | hf) * Dsz) + c) * 16 + j] = (f16)w1[i];
    }
    for (int i = tid; i < Dsz; i += 256) w1last[i] = (f16)w1[128 * Dsz + i];
    for (int i = tid; i < 16 * Dsz; i += 256) {
        int j = i >> 7, d = i & 127;
        X[j][d] = selfv[(size_t)(pbase + j) * Dsz + d];
    }
    for (int i = tid; i < 16 * Ssz; i += 256) wts[i] = nbw[(size_t)pbase * Ssz + i];
    __syncthreads();

    float w2v[8];
#pragma unroll
    for (int t = 0; t < 8; ++t) w2v[t] = w2[m16 + 16 * t];

    // each wave handles pairs jj = wv and wv+8 sequentially
    for (int rep = 0; rep < 2; ++rep) {
        const int jj = wv + rep * 8;
        const int p  = pbase + jj;

        // ---- neighbor tile -> LDS via CDNA5 async copy (ASYNCcnt path) ----
        {
            const char* gsrc = (const char*)(nbrv + (size_t)p * (Ssz * Dsz));
            unsigned ldsbase = (unsigned)(uintptr_t)(&nbuf[wv][0]);
#pragma unroll
            for (int it = 0; it < 4; ++it) {            // 128 chunks of 16B, 32 lanes
                const int i = it * 32 + lane;
                unsigned laddr = ldsbase + i * 16;
                const void* ga = gsrc + (size_t)i * 16;
                asm volatile("global_load_async_to_lds_b128 %0, %1, off"
                             :: "v"(laddr), "v"(ga) : "memory");
            }
            asm volatile("s_wait_asynccnt 0x0" ::: "memory");
        }

        // ---- stage 1: alpha1 = feat @ w1  (16x128x128 via f16 WMMA, f32 acc) ----
        v8f acc[8];
#pragma unroll
        for (int t = 0; t < 8; ++t) acc[t] = v8f{};

#pragma unroll
        for (int kc = 0; kc < 4; ++kc) {
            const int kbase = kc * 32;
            // A lane K pattern: two contiguous 8-float runs
            const int kA = kbase + (lo ? 0 : 8);
            const int kB = kbase + (lo ? 16 : 24);
            const float* xr = &X[jj][0];
            const float* nr = &nbuf[wv][m16 * Dsz];
            const float4 s0 = *(const float4*)&xr[kA];
            const float4 s1 = *(const float4*)&xr[kA + 4];
            const float4 s2 = *(const float4*)&xr[kB];
            const float4 s3 = *(const float4*)&xr[kB + 4];
            const float4 n0 = *(const float4*)&nr[kA];
            const float4 n1 = *(const float4*)&nr[kA + 4];
            const float4 n2 = *(const float4*)&nr[kB];
            const float4 n3 = *(const float4*)&nr[kB + 4];
            v16h a;
            a[0]  = (f16)(s0.x * n0.x); a[1]  = (f16)(s0.y * n0.y);
            a[2]  = (f16)(s0.z * n0.z); a[3]  = (f16)(s0.w * n0.w);
            a[4]  = (f16)(s1.x * n1.x); a[5]  = (f16)(s1.y * n1.y);
            a[6]  = (f16)(s1.z * n1.z); a[7]  = (f16)(s1.w * n1.w);
            a[8]  = (f16)(s2.x * n2.x); a[9]  = (f16)(s2.y * n2.y);
            a[10] = (f16)(s2.z * n2.z); a[11] = (f16)(s2.w * n2.w);
            a[12] = (f16)(s3.x * n3.x); a[13] = (f16)(s3.y * n3.y);
            a[14] = (f16)(s3.z * n3.z); a[15] = (f16)(s3.w * n3.w);
#pragma unroll
            for (int t = 0; t < 8; ++t) {
                const int col = m16 + 16 * t;
                const v16h b = *(const v16h*)&w1p[((((kc << 1) | hh) * Dsz) + col) * 16];
                acc[t] = __builtin_amdgcn_wmma_f32_16x16x32_f16(
                    false, a, false, b, (short)0, acc[t], false, false);
            }
        }

        // ---- +edge-weight column (w1 row 128), leaky-relu, dot with w2 ----
        float wrow[8], part[8];
#pragma unroll
        for (int r = 0; r < 8; ++r) {
            wrow[r] = wts[jj * Ssz + r + (lo ? 0 : 8)]; // row M = r (+8 for hi lanes)
            part[r] = 0.f;
        }
#pragma unroll
        for (int t = 0; t < 8; ++t) {
            const float w1l = (float)w1last[m16 + 16 * t];
#pragma unroll
            for (int r = 0; r < 8; ++r) {
                float x = acc[t][r] + wrow[r] * w1l;
                x = x > 0.f ? x : 0.2f * x;           // leaky_relu(0.2)
                part[r] += x * w2v[t];
            }
        }
        // reduce over N (16 lanes within each half)
#pragma unroll
        for (int off = 1; off < 16; off <<= 1)
#pragma unroll
            for (int r = 0; r < 8; ++r) part[r] += __shfl_xor(part[r], off, 32);

        // ---- softmax over the 16 rows (8 per half-wave) ----
        float mx = part[0];
#pragma unroll
        for (int r = 1; r < 8; ++r) mx = fmaxf(mx, part[r]);
        mx = fmaxf(mx, __shfl_xor(mx, 16, 32));
        float e[8], sum = 0.f;
#pragma unroll
        for (int r = 0; r < 8; ++r) { e[r] = __expf(part[r] - mx); sum += e[r]; }
        sum += __shfl_xor(sum, 16, 32);
        const float inv = 1.f / sum;
        float an[8];
#pragma unroll
        for (int r = 0; r < 8; ++r) an[r] = e[r] * inv;

        // broadcast the 16 alphas to all lanes (readlane, no LDS round-trip)
        float al[16];
#pragma unroll
        for (int m = 0; m < 8; ++m) al[m] = __shfl(an[m], 0, 32);
#pragma unroll
        for (int m = 0; m < 8; ++m) al[m + 8] = __shfl(an[m], 16, 32);

        // ---- nb = alpha^T @ neighbor (f32 from LDS) ----
        float4 nb4 = make_float4(0.f, 0.f, 0.f, 0.f);
#pragma unroll
        for (int m = 0; m < 16; ++m) {
            float4 v = ((const float4*)(nbuf[wv] + m * Dsz))[lane];
            nb4.x += al[m] * v.x; nb4.y += al[m] * v.y;
            nb4.z += al[m] * v.z; nb4.w += al[m] * v.w;
        }
        // second output (nb) + stash into concat buffer
        ((float4*)(out + (size_t)(Bsz * Nsz) * Dsz + (size_t)p * Dsz))[lane] = nb4;
        ((float4*)&X[jj][Dsz])[lane] = nb4;
    }

    // ---- stage 3: out = relu([self||nb] @ w3), 16x256x128, one 16x16 tile per wave ----
    __syncthreads();                       // all nb rows of X ready; nbuf free
    f16* w3p = (f16*)&nbuf[0][0];          // swizzled w3: 8 kc * 2 half * 128 col * 16 = 64 KB
    for (int i = tid; i < 256 * Dsz; i += 256) {
        int k = i >> 7, c = i & 127;
        int kc = k >> 5, rem = k & 31, hf = (rem >> 3) & 1, j = (rem & 7) | ((rem >> 4) << 3);
        w3p[((((kc << 1) | hf) * Dsz) + c) * 16 + j] = (f16)w3[i];
    }
    __syncthreads();

    v8f c = v8f{};
    const int col = m16 + 16 * wv;
#pragma unroll
    for (int kc = 0; kc < 8; ++kc) {
        const int kbase = kc * 32;
        const int kA = kbase + (lo ? 0 : 8);
        const int kB = kbase + (lo ? 16 : 24);
        const float* xr = &X[m16][0];
        const float4 s0 = *(const float4*)&xr[kA];
        const float4 s1 = *(const float4*)&xr[kA + 4];
        const float4 s2 = *(const float4*)&xr[kB];
        const float4 s3 = *(const float4*)&xr[kB + 4];
        v16h a;
        a[0]  = (f16)s0.x; a[1]  = (f16)s0.y; a[2]  = (f16)s0.z; a[3]  = (f16)s0.w;
        a[4]  = (f16)s1.x; a[5]  = (f16)s1.y; a[6]  = (f16)s1.z; a[7]  = (f16)s1.w;
        a[8]  = (f16)s2.x; a[9]  = (f16)s2.y; a[10] = (f16)s2.z; a[11] = (f16)s2.w;
        a[12] = (f16)s3.x; a[13] = (f16)s3.y; a[14] = (f16)s3.z; a[15] = (f16)s3.w;
        const v16h b = *(const v16h*)&w3p[((((kc << 1) | hh) * Dsz) + col) * 16];
        c = __builtin_amdgcn_wmma_f32_16x16x32_f16(
            false, a, false, b, (short)0, c, false, false);
    }
#pragma unroll
    for (int r = 0; r < 8; ++r) {
        const int M = r + (lo ? 0 : 8);    // pair row within block
        float v = c[r];
        v = v > 0.f ? v : 0.f;             // relu
        out[(size_t)(pbase + M) * Dsz + col] = v;
    }
}

extern "C" void kernel_launch(void* const* d_in, const int* in_sizes, int n_in,
                              void* d_out, int out_size, void* d_ws, size_t ws_size,
                              hipStream_t stream) {
    const float* selfv = (const float*)d_in[0];
    const float* nbrv  = (const float*)d_in[1];
    const float* nbw   = (const float*)d_in[2];
    // d_in[3] extra_vector, d_in[4] masks, d_in[5] batch_size: unused in math
    const float* w1    = (const float*)d_in[6];
    const float* w2    = (const float*)d_in[7];
    const float* w3    = (const float*)d_in[8];
    float* out = (float*)d_out;

    dim3 grid((Bsz * Nsz) / 16);   // 1024 blocks of 16 pairs
    dim3 block(256);               // 8 waves (wave32)
    hipLaunchKernelGGL(gagg_kernel, grid, block, 0, stream,
                       selfv, nbrv, nbw, w1, w2, w3, out);
}
